// GATNetwork_32461362823868
// MI455X (gfx1250) — matrix operand order repaired and checked by reading, not compiled, not run on hardware
//
#include <hip/hip_runtime.h>
#include <hip/hip_bf16.h>
#include <math.h>

typedef __attribute__((ext_vector_type(16))) _Float16 v16h;
typedef __attribute__((ext_vector_type(8)))  _Float16 v8h;
typedef __attribute__((ext_vector_type(4)))  _Float16 v4h;
typedef __attribute__((ext_vector_type(8)))  float    v8f;

#define NEG_SLOPE 0.2f
#define SOFT_EPS  1e-16f
#define AGGR_EPW  4   // edges per wave in aggregation pass

// ---------------- small utility kernels ----------------

__global__ void k_fill(float* __restrict__ p, float v, int n) {
    int i = blockIdx.x * blockDim.x + threadIdx.x;
    if (i < n) p[i] = v;
}

// Convert W (fp32, row-major [k=64][n=64]) into fp16 stored in WMMA B-fragment
// order: pos = ((kchunk*4 + ntile)*32 + lane)*16 + j, lane = 16*(krem/16) + (n%16),
// j = krem%16.  Each lane's fragment bytes become two contiguous 16B chunks.
__global__ void k_cvt_frag(const float* __restrict__ w, _Float16* __restrict__ wf) {
    int i = blockIdx.x * blockDim.x + threadIdx.x;   // i = k*64 + n
    if (i >= 4096) return;
    int k = i >> 6, n = i & 63;
    int chunk = k >> 5, kr = k & 31;
    int b = kr >> 4, j = kr & 15;
    int nt = n >> 4, nc = n & 15;
    int lane = b * 16 + nc;
    int pos = ((chunk * 4 + nt) * 32 + lane) * 16 + j;
    wf[pos] = (_Float16)w[i];
}

// ---------------- WMMA GEMM: H[nrows x 64] = X[nrows x 64] @ W[64 x 64] ----------------
// 128 threads = 4 wave32; each wave computes a 16x64 strip of the 64x64 block tile.
// A fragments: ds_load_b128 (contiguous k = 16b + s per lane, same order as Wf).
// B fragments: global_load_b128 pairs from pre-swizzled, L2-resident Wf.
// Whole-tile bounds check hoisted: full blocks take a branch-free load/store path.

__global__ __launch_bounds__(128) void k_gemm64(const float* __restrict__ X,
                                                const _Float16* __restrict__ Wf,
                                                float* __restrict__ Hout, int nrows) {
    __shared__ _Float16 sx[64 * 72];   // x tile, row-major [m][k], 144B row stride
    const int tid  = threadIdx.x;
    const int base = blockIdx.x * 64;
    const bool full = (base + 64 <= nrows);

    // stage x tile: 8 x (float4 load -> 4 halfs -> 8B LDS store) per thread
    if (full) {
#pragma unroll
        for (int ch = 0; ch < 8; ++ch) {
            int idx = ch * 512 + tid * 4;
            int r = idx >> 6, c = idx & 63;
            float4 xv = *(const float4*)(X + (size_t)(base + r) * 64 + c);
            v4h hv;
            hv[0] = (_Float16)xv.x; hv[1] = (_Float16)xv.y;
            hv[2] = (_Float16)xv.z; hv[3] = (_Float16)xv.w;
            *(v4h*)(&sx[r * 72 + c]) = hv;
        }
    } else {
#pragma unroll
        for (int ch = 0; ch < 8; ++ch) {
            int idx = ch * 512 + tid * 4;
            int r = idx >> 6, c = idx & 63;
            int row = base + r;
            float4 xv = make_float4(0.f, 0.f, 0.f, 0.f);
            if (row < nrows) xv = *(const float4*)(X + (size_t)row * 64 + c);
            v4h hv;
            hv[0] = (_Float16)xv.x; hv[1] = (_Float16)xv.y;
            hv[2] = (_Float16)xv.z; hv[3] = (_Float16)xv.w;
            *(v4h*)(&sx[r * 72 + c]) = hv;
        }
    }
    __syncthreads();

    const int wave = tid >> 5;
    const int lane = tid & 31;
    const int b    = lane >> 4;    // K-half selector within fragment
    const int q    = lane & 15;    // row (A) / column (B,C/D) index
    const int m0   = wave * 16;

    v8f acc[4] = {};
    const _Float16* arow = &sx[(m0 + q) * 72];

#pragma unroll
    for (int kc = 0; kc < 2; ++kc) {
        const int kk = kc * 32;
        // A fragment: lane(b,q) holds row m0+q, K in [kk+16b, +16)
        v8h alo = *(const v8h*)(arow + kk + 16 * b);
        v8h ahi = *(const v8h*)(arow + kk + 16 * b + 8);
        v16h a = __builtin_shufflevector(alo, ahi,
                                         0, 1, 2, 3, 4, 5, 6, 7,
                                         8, 9, 10, 11, 12, 13, 14, 15);
#pragma unroll
        for (int nt = 0; nt < 4; ++nt) {
            const _Float16* bp = Wf + (((size_t)kc * 4 + nt) * 32 + lane) * 16;
            v8h blo = *(const v8h*)(bp);
            v8h bhi = *(const v8h*)(bp + 8);
            v16h bm = __builtin_shufflevector(blo, bhi,
                                              0, 1, 2, 3, 4, 5, 6, 7,
                                              8, 9, 10, 11, 12, 13, 14, 15);
            acc[nt] = __builtin_amdgcn_wmma_f32_16x16x32_f16(
                false, a, false, bm, (short)0, acc[nt], false, false);
        }
    }

    // C/D layout: element r -> M = r + 8*b, N = q
    if (full) {
#pragma unroll
        for (int nt = 0; nt < 4; ++nt) {
#pragma unroll
            for (int r = 0; r < 8; ++r) {
                int row = base + m0 + r + 8 * b;
                Hout[(size_t)row * 64 + nt * 16 + q] = acc[nt][r];
            }
        }
    } else {
#pragma unroll
        for (int nt = 0; nt < 4; ++nt) {
#pragma unroll
            for (int r = 0; r < 8; ++r) {
                int row = base + m0 + r + 8 * b;
                if (row < nrows) Hout[(size_t)row * 64 + nt * 16 + q] = acc[nt][r];
            }
        }
    }
}

// ---------------- attention logits: a_src/a_dst[node,h] = sum_c h*att ----------------

__global__ void k_attn(const float* __restrict__ Hm, const float* __restrict__ att_s,
                       const float* __restrict__ att_d, float* __restrict__ as_,
                       float* __restrict__ ad_, int N, int NH, int C) {
    int i = blockIdx.x * blockDim.x + threadIdx.x;
    if (i >= N * NH) return;
    int node = i / NH, hh = i - node * NH;
    const float* hp = Hm + (size_t)node * NH * C + hh * C;
    float ss = 0.f, sd = 0.f;
    for (int c = 0; c < C; ++c) {
        float hv = hp[c];
        ss += hv * att_s[hh * C + c];
        sd += hv * att_d[hh * C + c];
    }
    as_[i] = ss;
    ad_[i] = sd;
}

// ---------------- edge softmax passes ----------------

__device__ __forceinline__ void atomicMaxF(float* addr, float v) {
    if (v >= 0.f) atomicMax((int*)addr, __float_as_int(v));
    else          atomicMin((unsigned int*)addr, __float_as_uint(v));
}

__device__ __forceinline__ void edge_endpoints(const int* __restrict__ ei, int E, int e,
                                               int& s, int& d) {
    if (e < E) { s = ei[e]; d = ei[E + e]; }
    else       { s = e - E; d = s; }       // appended self-loops
}

__global__ void k_edge_max(const int* __restrict__ ei, int E, int N,
                           const float* __restrict__ as_, const float* __restrict__ ad_,
                           float* __restrict__ amax, int NH) {
    int e = blockIdx.x * blockDim.x + threadIdx.x;
    if (e >= E + N) return;
    int s, d; edge_endpoints(ei, E, e, s, d);
    for (int hh = 0; hh < NH; ++hh) {
        float a = as_[s * NH + hh] + ad_[d * NH + hh];
        a = (a > 0.f) ? a : NEG_SLOPE * a;
        atomicMaxF(&amax[d * NH + hh], a);
    }
}

__global__ void k_edge_sum(const int* __restrict__ ei, int E, int N,
                           const float* __restrict__ as_, const float* __restrict__ ad_,
                           const float* __restrict__ amax, float* __restrict__ den, int NH) {
    int e = blockIdx.x * blockDim.x + threadIdx.x;
    if (e >= E + N) return;
    int s, d; edge_endpoints(ei, E, e, s, d);
    for (int hh = 0; hh < NH; ++hh) {
        float a = as_[s * NH + hh] + ad_[d * NH + hh];
        a = (a > 0.f) ? a : NEG_SLOPE * a;
        atomicAdd(&den[d * NH + hh], expf(a - amax[d * NH + hh]));
    }
}

// wave-per-edge aggregation, AGGR_EPW edges per wave with software prefetch of the
// random h[src] rows (global_prefetch_b8).  32 lanes cover 64 channels (2 each).
__global__ void k_edge_aggr(const int* __restrict__ ei, int E, int N,
                            const float* __restrict__ as_, const float* __restrict__ ad_,
                            const float* __restrict__ amax, const float* __restrict__ den,
                            const float* __restrict__ Hm, float* __restrict__ out,
                            int NH, int C) {
    int wv   = (blockIdx.x * blockDim.x + threadIdx.x) >> 5;
    int lane = threadIdx.x & 31;
    int e0   = wv * AGGR_EPW;
    int Etot = E + N;

    int ss[AGGR_EPW], dd[AGGR_EPW];
#pragma unroll
    for (int u = 0; u < AGGR_EPW; ++u) {
        int e = e0 + u;
        if (e < Etot) {
            edge_endpoints(ei, E, e, ss[u], dd[u]);
            __builtin_prefetch(&Hm[(size_t)ss[u] * 64 + lane], 0, 1);
            __builtin_prefetch(&Hm[(size_t)ss[u] * 64 + lane + 32], 0, 1);
        }
    }

#pragma unroll
    for (int u = 0; u < AGGR_EPW; ++u) {
        int e = e0 + u;
        if (e >= Etot) continue;
        int s = ss[u], d = dd[u];
#pragma unroll 2
        for (int c = lane; c < 64; c += 32) {
            int hh = c / C;
            float a = as_[s * NH + hh] + ad_[d * NH + hh];
            a = (a > 0.f) ? a : NEG_SLOPE * a;
            float coef = expf(a - amax[d * NH + hh]) / (den[d * NH + hh] + SOFT_EPS);
            atomicAdd(&out[(size_t)d * 64 + c], Hm[(size_t)s * 64 + c] * coef);
        }
    }
}

// ---------------- epilogues ----------------

__global__ void k_bias_elu(float* __restrict__ y, const float* __restrict__ bias, int n) {
    int i = blockIdx.x * blockDim.x + threadIdx.x;
    if (i >= n) return;
    float v = y[i] + bias[i & 63];
    y[i] = (v > 0.f) ? v : expm1f(v);
}

// wave per node: bias add + log_softmax over 64 channels
__global__ void k_bias_lsm(float* __restrict__ y, const float* __restrict__ bias, int N) {
    int gid  = blockIdx.x * blockDim.x + threadIdx.x;
    int node = gid >> 5;
    int lane = threadIdx.x & 31;
    if (node >= N) return;
    float* row = y + (size_t)node * 64;
    float z0 = row[lane]      + bias[lane];
    float z1 = row[lane + 32] + bias[lane + 32];
    float m = fmaxf(z0, z1);
    for (int o = 16; o > 0; o >>= 1) m = fmaxf(m, __shfl_xor(m, o, 32));
    float s = expf(z0 - m) + expf(z1 - m);
    for (int o = 16; o > 0; o >>= 1) s += __shfl_xor(s, o, 32);
    float l = m + logf(s);
    row[lane]      = z0 - l;
    row[lane + 32] = z1 - l;
}

// ---------------- host orchestration ----------------

extern "C" void kernel_launch(void* const* d_in, const int* in_sizes, int n_in,
                              void* d_out, int out_size, void* d_ws, size_t ws_size,
                              hipStream_t stream) {
    const float* x   = (const float*)d_in[0];
    const int*   ei  = (const int*)d_in[1];
    const float* W1  = (const float*)d_in[2];
    const float* as1 = (const float*)d_in[3];
    const float* ad1 = (const float*)d_in[4];
    const float* b1  = (const float*)d_in[5];
    const float* W2  = (const float*)d_in[6];
    const float* as2 = (const float*)d_in[7];
    const float* ad2 = (const float*)d_in[8];
    const float* b2  = (const float*)d_in[9];

    const int N    = 50000;
    const int E    = in_sizes[1] / 2;
    const int T    = in_sizes[0] / (N * 64);
    const int Etot = E + N;

    // workspace layout (floats)
    float* ws   = (float*)d_ws;
    float* h    = ws;                            // N*64
    float* y1   = h    + (size_t)N * 64;         // N*64
    float* asrc = y1   + (size_t)N * 64;         // N*8 (max heads)
    float* adst = asrc + (size_t)N * 8;
    float* amax = adst + (size_t)N * 8;
    float* den  = amax + (size_t)N * 8;
    _Float16* w1f = (_Float16*)(den + (size_t)N * 8);  // 4096 halfs, fragment order
    _Float16* w2f = w1f + 4096;

    const int B = 256;
    auto cdiv = [](long long a, long long b) { return (int)((a + b - 1) / b); };

    k_cvt_frag<<<cdiv(4096, B), B, 0, stream>>>(W1, w1f);
    k_cvt_frag<<<cdiv(4096, B), B, 0, stream>>>(W2, w2f);

    const int gemm_blocks = cdiv(N, 64);
    const int edge_blocks = cdiv(Etot, B);
    const int aggr_blocks = cdiv((long long)cdiv(Etot, AGGR_EPW) * 32, B);

    for (int t = 0; t < T; ++t) {
        const float* xt  = x + (size_t)t * N * 64;
        float* outt = (float*)d_out + (size_t)t * N * 64;

        // ---- layer 1: GATConv(64 -> 8 heads x 8), ELU ----
        k_fill<<<cdiv(N * 8, B),  B, 0, stream>>>(amax, -INFINITY, N * 8);
        k_fill<<<cdiv(N * 8, B),  B, 0, stream>>>(den, 0.f, N * 8);
        k_fill<<<cdiv(N * 64, B), B, 0, stream>>>(y1, 0.f, N * 64);
        k_gemm64<<<gemm_blocks, 128, 0, stream>>>(xt, w1f, h, N);
        k_attn<<<cdiv(N * 8, B), B, 0, stream>>>(h, as1, ad1, asrc, adst, N, 8, 8);
        k_edge_max<<<edge_blocks, B, 0, stream>>>(ei, E, N, asrc, adst, amax, 8);
        k_edge_sum<<<edge_blocks, B, 0, stream>>>(ei, E, N, asrc, adst, amax, den, 8);
        k_edge_aggr<<<aggr_blocks, B, 0, stream>>>(ei, E, N, asrc, adst, amax, den, h, y1, 8, 8);
        k_bias_elu<<<cdiv(N * 64, B), B, 0, stream>>>(y1, b1, N * 64);

        // ---- layer 2: GATConv(64 -> 1 head x 64), log_softmax ----
        k_fill<<<cdiv(N, B),      B, 0, stream>>>(amax, -INFINITY, N);
        k_fill<<<cdiv(N, B),      B, 0, stream>>>(den, 0.f, N);
        k_fill<<<cdiv(N * 64, B), B, 0, stream>>>(outt, 0.f, N * 64);
        k_gemm64<<<gemm_blocks, 128, 0, stream>>>(y1, w2f, h, N);
        k_attn<<<cdiv(N, B), B, 0, stream>>>(h, as2, ad2, asrc, adst, N, 1, 64);
        k_edge_max<<<edge_blocks, B, 0, stream>>>(ei, E, N, asrc, adst, amax, 1);
        k_edge_sum<<<edge_blocks, B, 0, stream>>>(ei, E, N, asrc, adst, amax, den, 1);
        k_edge_aggr<<<aggr_blocks, B, 0, stream>>>(ei, E, N, asrc, adst, amax, den, h, outt, 1, 64);
        k_bias_lsm<<<cdiv((long long)N * 32, B), B, 0, stream>>>(outt, b2, N);
    }
}
